// DeepSSM_44375602102817
// MI455X (gfx1250) — matrix operand order repaired and checked by reading, not compiled, not run on hardware
//
#include <hip/hip_runtime.h>
#include <hip/hip_bf16.h>
#include <math.h>

typedef __attribute__((ext_vector_type(16))) _Float16 v16h;
typedef __attribute__((ext_vector_type(4)))  _Float16 v4h;
typedef __attribute__((ext_vector_type(8)))  float    v8f;
typedef __attribute__((ext_vector_type(4)))  float    v4f;

#define D_MODEL 512
#define D_FF    2048
#define N_HEADS 8
#define EDIM    64
#define BATCH   32
#define SEQLEN  512
#define NTOK    (BATCH * SEQLEN)   // 16384 tokens for enc and for dec
#define PRED    256
#define C_OUT   7
#define H_DIM   32

// ---------------------------------------------------------------------------
// WMMA fragment helpers (f16 16x16x32, wave32; A/B in this file's contiguous k order, C/D per ISA)
// A (16xK=32): lane holds row m=lane%16; element e -> K = e + 16*(lane/16)
// B (K=32x16): lane holds col n=lane%16; element e -> K = e + 16*(lane/16)
// C/D (16x16 f32): lane holds col n=lane%16; VGPR r -> row m = r + 8*(lane/16)
// ---------------------------------------------------------------------------
__device__ __forceinline__ v16h frag_a_lds(const _Float16* base, int ld) {
  const int lane = threadIdx.x & 31;
  const int m  = lane & 15;
  const int kb = (lane >> 4) * 16;
  const _Float16* p = base + m * ld;
  v4h f0 = *(const v4h*)(p + kb);
  v4h f1 = *(const v4h*)(p + kb + 4);
  v4h f2 = *(const v4h*)(p + kb + 8);
  v4h f3 = *(const v4h*)(p + kb + 12);
  v16h a;
#pragma unroll
  for (int j = 0; j < 4; ++j) {
    a[j] = f0[j]; a[4 + j] = f1[j]; a[8 + j] = f2[j]; a[12 + j] = f3[j];
  }
  return a;
}

__device__ __forceinline__ v16h frag_b_lds(const _Float16* base, int ld) {
  const int lane = threadIdx.x & 31;
  const int n  = lane & 15;
  const int kb = (lane >> 4) * 16;
  v16h b;
#pragma unroll
  for (int e = 0; e < 16; ++e) b[e] = base[(kb + e) * ld + n];
  return b;
}

__device__ __forceinline__ float gelu_tanh(float x) {
  float x3 = x * x * x;
  return 0.5f * x * (1.0f + tanhf(0.7978845608028654f * (x + 0.044715f * x3)));
}

// ---------------------------------------------------------------------------
// Generic GEMM: C[M,N] = op(A[M,K] @ W[K,N] + bias) (+res)
// mode 0: +bias ; mode 1: gelu(+bias) ; mode 2: +bias +res
// Block tile 64x64, BK=64, 256 threads = 8 waves, each wave -> 16x32 via 4 WMMA.
// Staging: each thread moves 4 x float4 (global_load_b128) per tile, packs to
// f16 quads in LDS. Next K-tile is prefetched (global_prefetch_b8).
// M, N, K must be multiples of 64 (true for all call sites).
// ---------------------------------------------------------------------------
__global__ __launch_bounds__(256) void gemm_kernel(
    const float* __restrict__ A, const float* __restrict__ W,
    const float* __restrict__ bias, const float* __restrict__ res,
    float* __restrict__ C, int M, int N, int K, int mode)
{
  __shared__ _Float16 As[64][72];   // 64 x 64 (+pad)
  __shared__ _Float16 Bs[64][72];   // 64 x 64 (+pad)

  const int tid  = threadIdx.x;
  const int wave = tid >> 5;
  const int lane = tid & 31;
  const int hh   = lane >> 4;
  const int nloc = lane & 15;
  const int m0 = blockIdx.y * 64;
  const int n0 = blockIdx.x * 64;
  const int subM = wave & 3;    // 4 row tiles of 16
  const int subN = wave >> 2;   // 2 col halves of 32

  const int srow = tid >> 4;    // staging row base (0..15), +16 per chunk
  const int sc4  = tid & 15;    // staging float4 column (0..15)

  v8f c0, c1;
#pragma unroll
  for (int r = 0; r < 8; ++r) { c0[r] = 0.0f; c1[r] = 0.0f; }

  for (int k0 = 0; k0 < K; k0 += 64) {
    if (k0 + 64 < K) {   // pull next K-tile toward L2/WGP$ while staging this one
      __builtin_prefetch(A + (size_t)(m0 + srow) * K + (k0 + 64) + sc4 * 4, 0, 0);
      __builtin_prefetch(W + (size_t)(k0 + 64 + srow) * N + n0 + sc4 * 4, 0, 0);
    }
    // stage A tile (64x64) and W tile (64x64) into LDS as f16 (4 float4 each)
#pragma unroll
    for (int p = 0; p < 4; ++p) {
      const int r = srow + p * 16;
      v4f fa = *(const v4f*)(A + (size_t)(m0 + r) * K + k0 + sc4 * 4);
      v4f fb = *(const v4f*)(W + (size_t)(k0 + r) * N + n0 + sc4 * 4);
      v4h ha, hb;
#pragma unroll
      for (int j = 0; j < 4; ++j) { ha[j] = (_Float16)fa[j]; hb[j] = (_Float16)fb[j]; }
      *(v4h*)&As[r][sc4 * 4] = ha;
      *(v4h*)&Bs[r][sc4 * 4] = hb;
    }
    __syncthreads();

    v16h a0  = frag_a_lds(&As[subM * 16][0],  72);
    v16h a1  = frag_a_lds(&As[subM * 16][32], 72);
    v16h b00 = frag_b_lds(&Bs[0][subN * 32],       72);
    v16h b01 = frag_b_lds(&Bs[0][subN * 32 + 16],  72);
    v16h b10 = frag_b_lds(&Bs[32][subN * 32],      72);
    v16h b11 = frag_b_lds(&Bs[32][subN * 32 + 16], 72);
    c0 = __builtin_amdgcn_wmma_f32_16x16x32_f16(false, a0, false, b00, (short)0, c0, false, false);
    c1 = __builtin_amdgcn_wmma_f32_16x16x32_f16(false, a0, false, b01, (short)0, c1, false, false);
    c0 = __builtin_amdgcn_wmma_f32_16x16x32_f16(false, a1, false, b10, (short)0, c0, false, false);
    c1 = __builtin_amdgcn_wmma_f32_16x16x32_f16(false, a1, false, b11, (short)0, c1, false, false);
    __syncthreads();
  }

  const int mbase = m0 + subM * 16 + 8 * hh;
#pragma unroll
  for (int f = 0; f < 2; ++f) {
    const int col = n0 + subN * 32 + f * 16 + nloc;
    const float bv = bias ? bias[col] : 0.0f;
#pragma unroll
    for (int r = 0; r < 8; ++r) {
      size_t idx = (size_t)(mbase + r) * N + col;
      float v = (f ? c1[r] : c0[r]) + bv;
      if (mode == 1) v = gelu_tanh(v);
      if (mode == 2) v += res[idx];
      C[idx] = v;
    }
  }
}

// ---------------------------------------------------------------------------
// Flash attention: block = 128 threads (4 waves), each wave owns a 16-row Q
// tile, streams S in 32-wide tiles. grid = (Lq/64, B*H).
// Q/K/V/O layout: (B, L, D_MODEL) with head h at cols [h*64, h*64+64).
// ---------------------------------------------------------------------------
__global__ __launch_bounds__(128) void attn_kernel(
    const float* __restrict__ Qg, const float* __restrict__ Kg,
    const float* __restrict__ Vg, float* __restrict__ Og,
    int Lq, int S, int causal)
{
  __shared__ _Float16 Pp[4][16][36];   // per-wave P patch (16x32 + pad)

  const int wave = threadIdx.x >> 5;
  const int lane = threadIdx.x & 31;
  const int hh   = lane >> 4;
  const int nloc = lane & 15;
  const int bh = blockIdx.y;
  const int b = bh / N_HEADS, h = bh % N_HEADS;
  const int q0 = blockIdx.x * 64 + wave * 16;

  const float* Qp = Qg + (size_t)b * Lq * D_MODEL + h * EDIM;
  const float* Kp = Kg + (size_t)b * S  * D_MODEL + h * EDIM;
  const float* Vp = Vg + (size_t)b * S  * D_MODEL + h * EDIM;

  // preload the Q A-fragments (K-dim 0..31 and 32..63), vectorized b128 loads
  v16h aq[2];
  {
    const int m  = nloc;
    const int kb = hh * 16;
    const float* qrow = Qp + (size_t)(q0 + m) * D_MODEL;
#pragma unroll
    for (int c = 0; c < 2; ++c) {
      v4f f0 = *(const v4f*)(qrow + c * 32 + kb);
      v4f f1 = *(const v4f*)(qrow + c * 32 + kb + 4);
      v4f f2 = *(const v4f*)(qrow + c * 32 + kb + 8);
      v4f f3 = *(const v4f*)(qrow + c * 32 + kb + 12);
#pragma unroll
      for (int j = 0; j < 4; ++j) {
        aq[c][j]      = (_Float16)f0[j];
        aq[c][4 + j]  = (_Float16)f1[j];
        aq[c][8 + j]  = (_Float16)f2[j];
        aq[c][12 + j] = (_Float16)f3[j];
      }
    }
  }

  float mstat[8], lstat[8];
  v8f o[4];
#pragma unroll
  for (int r = 0; r < 8; ++r) { mstat[r] = -3.0e38f; lstat[r] = 0.0f; }
#pragma unroll
  for (int j = 0; j < 4; ++j)
#pragma unroll
    for (int r = 0; r < 8; ++r) o[j][r] = 0.0f;

  const int smax = causal ? ((q0 + 16) < S ? (q0 + 16) : S) : S;
  const float scale = 0.125f;   // 1/sqrt(64)

  for (int s0 = 0; s0 < smax; s0 += 32) {
    // ---- scores (16x32) = Q(16x64) @ K^T(64x32) : 4 WMMAs ----
    v8f sc[2];
#pragma unroll
    for (int t2 = 0; t2 < 2; ++t2) {
#pragma unroll
      for (int r = 0; r < 8; ++r) sc[t2][r] = 0.0f;
      const int srow = s0 + t2 * 16 + nloc;
      const int kb2  = hh * 16;
#pragma unroll
      for (int kc = 0; kc < 2; ++kc) {
        const float* krow = Kp + (size_t)srow * D_MODEL + kc * 32 + kb2;
        v4f k0v = *(const v4f*)(krow);
        v4f k1v = *(const v4f*)(krow + 4);
        v4f k2v = *(const v4f*)(krow + 8);
        v4f k3v = *(const v4f*)(krow + 12);
        v16h bk;
#pragma unroll
        for (int j = 0; j < 4; ++j) {
          bk[j]      = (_Float16)k0v[j];
          bk[4 + j]  = (_Float16)k1v[j];
          bk[8 + j]  = (_Float16)k2v[j];
          bk[12 + j] = (_Float16)k3v[j];
        }
        sc[t2] = __builtin_amdgcn_wmma_f32_16x16x32_f16(false, aq[kc], false, bk, (short)0, sc[t2], false, false);
      }
    }
    // ---- scale + causal mask + online softmax ----
#pragma unroll
    for (int t2 = 0; t2 < 2; ++t2) {
      const int col = s0 + t2 * 16 + nloc;
#pragma unroll
      for (int r = 0; r < 8; ++r) {
        const int row = q0 + r + 8 * hh;
        float v = sc[t2][r] * scale;
        if (causal && col > row) v = -3.0e38f;
        sc[t2][r] = v;
      }
    }
#pragma unroll
    for (int r = 0; r < 8; ++r) {
      float rm = fmaxf(sc[0][r], sc[1][r]);
      rm = fmaxf(rm, __shfl_xor(rm, 8));
      rm = fmaxf(rm, __shfl_xor(rm, 4));
      rm = fmaxf(rm, __shfl_xor(rm, 2));
      rm = fmaxf(rm, __shfl_xor(rm, 1));
      float mnew = fmaxf(mstat[r], rm);
      float corr = expf(mstat[r] - mnew);
      float p0 = expf(sc[0][r] - mnew);
      float p1 = expf(sc[1][r] - mnew);
      sc[0][r] = p0; sc[1][r] = p1;
      float rs = p0 + p1;
      rs += __shfl_xor(rs, 8);
      rs += __shfl_xor(rs, 4);
      rs += __shfl_xor(rs, 2);
      rs += __shfl_xor(rs, 1);
      lstat[r] = lstat[r] * corr + rs;
      mstat[r] = mnew;
#pragma unroll
      for (int j = 0; j < 4; ++j) o[j][r] *= corr;
    }
    // ---- C-layout P -> A-layout via wave-private LDS patch ----
#pragma unroll
    for (int t2 = 0; t2 < 2; ++t2)
#pragma unroll
      for (int r = 0; r < 8; ++r)
        Pp[wave][r + 8 * hh][t2 * 16 + nloc] = (_Float16)sc[t2][r];
    asm volatile("s_wait_dscnt 0x0" ::: "memory");   // wave-local DS RAW fence
    v16h ap;
    {
      const int m  = nloc;
      const int kb = hh * 16;
      v4h p0 = *(const v4h*)&Pp[wave][m][kb];
      v4h p1 = *(const v4h*)&Pp[wave][m][kb + 4];
      v4h p2 = *(const v4h*)&Pp[wave][m][kb + 8];
      v4h p3 = *(const v4h*)&Pp[wave][m][kb + 12];
#pragma unroll
      for (int j = 0; j < 4; ++j) {
        ap[j] = p0[j]; ap[4 + j] = p1[j]; ap[8 + j] = p2[j]; ap[12 + j] = p3[j];
      }
    }
    // ---- O += P(16x32) @ V(32x64) : 4 WMMAs ----
    const int kb2 = hh * 16;
#pragma unroll
    for (int j = 0; j < 4; ++j) {
      v16h bv;
#pragma unroll
      for (int e = 0; e < 16; ++e)
        bv[e] = (_Float16)Vp[(size_t)(s0 + kb2 + e) * D_MODEL + j * 16 + nloc];
      o[j] = __builtin_amdgcn_wmma_f32_16x16x32_f16(false, ap, false, bv, (short)0, o[j], false, false);
    }
  }

  // ---- normalize + write (b, q0+m, h*64 + 16j + n) ----
  float inv[8];
#pragma unroll
  for (int r = 0; r < 8; ++r) inv[r] = 1.0f / lstat[r];
#pragma unroll
  for (int j = 0; j < 4; ++j)
#pragma unroll
    for (int r = 0; r < 8; ++r) {
      const int row = q0 + r + 8 * hh;
      Og[(size_t)(b * Lq + row) * D_MODEL + h * EDIM + j * 16 + nloc] = o[j][r] * inv[r];
    }
}

// ---------------------------------------------------------------------------
// LayerNorm over last dim (512), one block per row; y = (x-m)/sqrt(v+eps)*g+b
// ---------------------------------------------------------------------------
__global__ __launch_bounds__(256) void ln_kernel(
    const float* __restrict__ x, const float* __restrict__ g,
    const float* __restrict__ b, float* __restrict__ y)
{
  __shared__ float red[256];
  const size_t row = blockIdx.x;
  const int tid = threadIdx.x;
  const float* xr = x + row * D_MODEL;
  float v0 = xr[tid], v1 = xr[tid + 256];
  red[tid] = v0 + v1;
  __syncthreads();
  for (int s = 128; s > 0; s >>= 1) { if (tid < s) red[tid] += red[tid + s]; __syncthreads(); }
  float mean = red[0] * (1.0f / 512.0f);
  __syncthreads();
  float d0 = v0 - mean, d1 = v1 - mean;
  red[tid] = d0 * d0 + d1 * d1;
  __syncthreads();
  for (int s = 128; s > 0; s >>= 1) { if (tid < s) red[tid] += red[tid + s]; __syncthreads(); }
  float rstd = rsqrtf(red[0] * (1.0f / 512.0f) + 1e-5f);
  y[row * D_MODEL + tid]       = d0 * rstd * g[tid]       + b[tid];
  y[row * D_MODEL + tid + 256] = d1 * rstd * g[tid + 256] + b[tid + 256];
}

// ---------------------------------------------------------------------------
// Embedding: out[b,t,:] = sum_k x[wrap(t-1+k)] @ conv[k] + PE(t,:) + mark @ tw
// grid (L, B), block 256, each thread 2 channels
// ---------------------------------------------------------------------------
__global__ __launch_bounds__(256) void emb_kernel(
    const float* __restrict__ x, const float* __restrict__ mark,
    const float* __restrict__ conv, const float* __restrict__ tw,
    float* __restrict__ out, int L)
{
  __shared__ float xs[21];
  __shared__ float ms[4];
  const int t = blockIdx.x, b = blockIdx.y;
  const int tid = threadIdx.x;
  const int tm1 = (t == 0) ? (L - 1) : (t - 1);
  const int tp1 = (t == L - 1) ? 0 : (t + 1);
  if (tid < 7) {
    xs[tid]      = x[((size_t)b * L + tm1) * 7 + tid];
    xs[7 + tid]  = x[((size_t)b * L + t)   * 7 + tid];
    xs[14 + tid] = x[((size_t)b * L + tp1) * 7 + tid];
  }
  if (tid < 4) ms[tid] = mark[((size_t)b * L + t) * 4 + tid];
  __syncthreads();

#pragma unroll
  for (int half = 0; half < 2; ++half) {
    const int d = tid + half * 256;
    float acc = 0.0f;
#pragma unroll
    for (int f = 0; f < 7; ++f) {
      acc += xs[f]      * conv[0 * 7 * D_MODEL + f * D_MODEL + d];
      acc += xs[7 + f]  * conv[1 * 7 * D_MODEL + f * D_MODEL + d];
      acc += xs[14 + f] * conv[2 * 7 * D_MODEL + f * D_MODEL + d];
    }
#pragma unroll
    for (int j = 0; j < 4; ++j) acc += ms[j] * tw[j * D_MODEL + d];
    const float div = expf(-(float)(2 * (d >> 1)) * (9.210340371976184f / 512.0f));
    const float arg = (float)t * div;
    acc += (d & 1) ? cosf(arg) : sinf(arg);
    out[((size_t)b * L + t) * D_MODEL + d] = acc;
  }
}

// ---------------------------------------------------------------------------
// out7[b,t,c] = decf[b, 256+t, :] @ proj.w[:,c] + proj.b[c]   (t in [0,256))
// ---------------------------------------------------------------------------
__global__ __launch_bounds__(256) void proj7_kernel(
    const float* __restrict__ decf, const float* __restrict__ Wp,
    const float* __restrict__ bp, float* __restrict__ o7)
{
  const int idx = blockIdx.x * blockDim.x + threadIdx.x;
  if (idx >= BATCH * PRED * C_OUT) return;
  const int c = idx % C_OUT;
  const int t = (idx / C_OUT) % PRED;
  const int b = idx / (C_OUT * PRED);
  const float* xr = decf + ((size_t)b * SEQLEN + PRED + t) * D_MODEL;
  float acc = bp[c];
  for (int d = 0; d < D_MODEL; ++d) acc += xr[d] * Wp[d * C_OUT + c];
  o7[idx] = acc;
}

// ---------------------------------------------------------------------------
// SSM scan: one wave32 per batch. Head weights staged in LDS; F/H/b/w/v are
// recomputed from out7 each step (7-term dots), l carried in registers+LDS.
// ---------------------------------------------------------------------------
__global__ __launch_bounds__(32) void scan_kernel(
    const float* __restrict__ o7,
    const float* __restrict__ WF, const float* __restrict__ bF,
    const float* __restrict__ WH, const float* __restrict__ bH,
    const float* __restrict__ Wb, const float* __restrict__ bb,
    const float* __restrict__ Ww, const float* __restrict__ bw,
    const float* __restrict__ Wv, const float* __restrict__ bv,
    const float* __restrict__ l0, const float* __restrict__ noise_e,
    float* __restrict__ out)
{
  __shared__ float sWF[7 * 1024];
  __shared__ float sbF[1024];
  __shared__ float sWH[7 * 224];
  __shared__ float sbH[224];
  __shared__ float sWb[49], sbb[7];
  __shared__ float sWw[224], sbw[32];
  __shared__ float sWv[49], sbv[7];
  __shared__ float lsh[32];

  const int b = blockIdx.x, i = threadIdx.x;
  for (int k = i; k < 7 * 1024; k += 32) sWF[k] = WF[k];
  for (int k = i; k < 1024;     k += 32) sbF[k] = bF[k];
  for (int k = i; k < 7 * 224;  k += 32) sWH[k] = WH[k];
  for (int k = i; k < 224;      k += 32) sbH[k] = bH[k];
  for (int k = i; k < 49;       k += 32) sWb[k] = Wb[k];
  for (int k = i; k < 7;        k += 32) sbb[k] = bb[k];
  for (int k = i; k < 224;      k += 32) sWw[k] = Ww[k];
  for (int k = i; k < 32;       k += 32) sbw[k] = bw[k];
  for (int k = i; k < 49;       k += 32) sWv[k] = Wv[k];
  for (int k = i; k < 7;        k += 32) sbv[k] = bv[k];

  float li = l0[b * H_DIM + i];
  const float e = noise_e[0];
  __syncthreads();

  for (int t = 0; t < PRED; ++t) {
    const float* o7r = o7 + ((size_t)b * PRED + t) * C_OUT;
    float oc[7];
#pragma unroll
    for (int c = 0; c < 7; ++c) oc[c] = o7r[c];

    __syncthreads();
    lsh[i] = li;
    __syncthreads();

    float acc = 0.0f;
    for (int j = 0; j < H_DIM; ++j) {
      float F = sbF[i * H_DIM + j];
#pragma unroll
      for (int c = 0; c < 7; ++c) F += oc[c] * sWF[c * 1024 + i * H_DIM + j];
      acc += F * lsh[j];
    }
    float wi = sbw[i];
#pragma unroll
    for (int c = 0; c < 7; ++c) wi += oc[c] * sWw[c * H_DIM + i];
    const float newl = acc + wi * e;

    __syncthreads();
    lsh[i] = newl;
    __syncthreads();

    if (i < C_OUT) {
      float z = sbb[i];
      float vi = sbv[i];
#pragma unroll
      for (int c = 0; c < 7; ++c) { z += oc[c] * sWb[c * 7 + i]; vi += oc[c] * sWv[c * 7 + i]; }
      for (int j = 0; j < H_DIM; ++j) {
        float H = sbH[i * H_DIM + j];
#pragma unroll
        for (int c = 0; c < 7; ++c) H += oc[c] * sWH[c * 224 + i * H_DIM + j];
        z += H * lsh[j];
      }
      z += vi * e;
      out[((size_t)b * PRED + t) * C_OUT + i] = z;
    }
    li = newl;
  }
}

// ---------------------------------------------------------------------------
// Host orchestration. Param leaf indices: setup_inputs() insertion order,
// recursively flattened:
//  0 x_enc, 1 x_mark_enc, 2 x_dec, 3 x_mark_dec, 4 noise_l0, 5 noise_e
//  6/7 enc_emb conv/tw, 8/9 dec_emb conv/tw
//  enc layer (base 10, 26): wq,bq,wk,bk,wv,bv,wo,bo, ln1.g,b, ln2.g,b, w1,b1,w2,b2
//  42/43 enc_norm g/b
//  dec layer (base 44): sa(8), ca(8), ln1(2), ln2(2), ln3(2), w1,b1,w2,b2
//  70/71 dec_norm, 72/73 proj, 74..83 pF,pH,pb,pw,pv (w,b each)
// ---------------------------------------------------------------------------
extern "C" void kernel_launch(void* const* d_in, const int* in_sizes, int n_in,
                              void* d_out, int out_size, void* d_ws, size_t ws_size,
                              hipStream_t stream)
{
  (void)in_sizes; (void)n_in; (void)out_size; (void)ws_size;
  auto in = [&](int i) { return (const float*)d_in[i]; };

  float* ws = (float*)d_ws;
  const size_t SZ = (size_t)NTOK * D_MODEL;   // 8.39M floats
  float* X    = ws;
  float* Qb   = X    + SZ;
  float* Kb   = Qb   + SZ;
  float* Vb   = Kb   + SZ;
  float* AO   = Vb   + SZ;
  float* TMP  = AO   + SZ;
  float* ENCF = TMP  + SZ;
  float* O7   = ENCF + SZ;                    // BATCH*PRED*C_OUT
  float* HID  = Qb;                           // aliases Qb..AO (16384 x 2048); disjoint in time

  auto gemm = [&](const float* A, const float* W, const float* bias, const float* res,
                  float* C, int M, int N, int K, int mode) {
    dim3 g(N / 64, M / 64);
    gemm_kernel<<<g, 256, 0, stream>>>(A, W, bias, res, C, M, N, K, mode);
  };
  auto ln = [&](const float* x, int gi, float* y) {
    ln_kernel<<<NTOK, 256, 0, stream>>>(x, in(gi), in(gi + 1), y);
  };
  auto attn = [&](const float* Q, const float* K, const float* V, float* O, int causal) {
    dim3 g(SEQLEN / 64, BATCH * N_HEADS);
    attn_kernel<<<g, 128, 0, stream>>>(Q, K, V, O, SEQLEN, SEQLEN, causal);
  };

  // ---------------- encoder ----------------
  emb_kernel<<<dim3(SEQLEN, BATCH), 256, 0, stream>>>(in(0), in(1), in(6), in(7), X, SEQLEN);
  for (int l = 0; l < 2; ++l) {
    const int p = (l == 0) ? 10 : 26;
    gemm(X, in(p + 0), in(p + 1), nullptr, Qb, NTOK, D_MODEL, D_MODEL, 0);
    gemm(X, in(p + 2), in(p + 3), nullptr, Kb, NTOK, D_MODEL, D_MODEL, 0);
    gemm(X, in(p + 4), in(p + 5), nullptr, Vb, NTOK, D_MODEL, D_MODEL, 0);
    attn(Qb, Kb, Vb, AO, 0);
    gemm(AO, in(p + 6), in(p + 7), X, TMP, NTOK, D_MODEL, D_MODEL, 2);
    ln(TMP, p + 8, X);
    gemm(X, in(p + 12), in(p + 13), nullptr, HID, NTOK, D_FF, D_MODEL, 1);
    gemm(HID, in(p + 14), in(p + 15), X, TMP, NTOK, D_MODEL, D_FF, 2);
    ln(TMP, p + 10, X);
  }
  ln(X, 42, ENCF);

  // ---------------- decoder ----------------
  emb_kernel<<<dim3(SEQLEN, BATCH), 256, 0, stream>>>(in(2), in(3), in(8), in(9), X, SEQLEN);
  {
    const int p = 44;
    // causal self-attention
    gemm(X, in(p + 0), in(p + 1), nullptr, Qb, NTOK, D_MODEL, D_MODEL, 0);
    gemm(X, in(p + 2), in(p + 3), nullptr, Kb, NTOK, D_MODEL, D_MODEL, 0);
    gemm(X, in(p + 4), in(p + 5), nullptr, Vb, NTOK, D_MODEL, D_MODEL, 0);
    attn(Qb, Kb, Vb, AO, 1);
    gemm(AO, in(p + 6), in(p + 7), X, TMP, NTOK, D_MODEL, D_MODEL, 2);
    ln(TMP, p + 16, X);
    // cross-attention against encoder output
    gemm(X,    in(p + 8),  in(p + 9),  nullptr, Qb, NTOK, D_MODEL, D_MODEL, 0);
    gemm(ENCF, in(p + 10), in(p + 11), nullptr, Kb, NTOK, D_MODEL, D_MODEL, 0);
    gemm(ENCF, in(p + 12), in(p + 13), nullptr, Vb, NTOK, D_MODEL, D_MODEL, 0);
    attn(Qb, Kb, Vb, AO, 0);
    gemm(AO, in(p + 14), in(p + 15), X, TMP, NTOK, D_MODEL, D_MODEL, 2);
    ln(TMP, p + 18, X);
    // FFN
    gemm(X, in(p + 22), in(p + 23), nullptr, HID, NTOK, D_FF, D_MODEL, 1);
    gemm(HID, in(p + 24), in(p + 25), X, TMP, NTOK, D_MODEL, D_FF, 2);
    ln(TMP, p + 20, X);
  }
  ln(X, 70, TMP);   // dec_norm -> TMP holds final decoder states

  // ---------------- head + scan ----------------
  proj7_kernel<<<(BATCH * PRED * C_OUT + 255) / 256, 256, 0, stream>>>(TMP, in(72), in(73), O7);
  scan_kernel<<<BATCH, 32, 0, stream>>>(O7,
      in(74), in(75), in(76), in(77), in(78), in(79),
      in(80), in(81), in(82), in(83), in(4), in(5), (float*)d_out);
}
